// NodeModel_79688823210547
// MI455X (gfx1250) — matrix operand order repaired and checked by reading, not compiled, not run on hardware
//
#include <hip/hip_runtime.h>
#include <hip/hip_bf16.h>

typedef __attribute__((ext_vector_type(16))) __bf16 v16bf;
typedef __attribute__((ext_vector_type(8)))  float  v8f;

union FragU { unsigned short u[16]; v16bf v; };

__device__ __forceinline__ unsigned short f2bf(float f) {
    unsigned int u = __float_as_uint(f);
    unsigned int r = (u + 0x7FFFu + ((u >> 16) & 1u)) >> 16;   // round-to-nearest-even
    return (unsigned short)r;
}

__device__ __forceinline__ unsigned int pack2bf(float a, float b) {
    return (unsigned int)f2bf(a) | ((unsigned int)f2bf(b) << 16);
}

__device__ __forceinline__ float elu1(float x) {
    return x > 0.0f ? x : (__expf(x) - 1.0f);
}

// A fragment (16x32 bf16, M x K) from an LDS row-major tile.
// Lanes 0-15: row m=lane, elems 0..7 = K[ko..ko+7],  elems 8..15 = K[ko+16..ko+23]
// Lanes 16-31: row m=lane-16, elems 0..7 = K[ko+8..], elems 8..15 = K[ko+24..]
__device__ __forceinline__ v16bf load_a_lds(const unsigned short* base, int stride,
                                            int ko, int lane) {
    const int m    = lane & 15;
    const int half = lane >> 4;
    const unsigned short* r = base + m * stride + ko + half * 8;
    FragU f;
#pragma unroll
    for (int e = 0; e < 8; ++e) { f.u[e] = r[e]; f.u[8 + e] = r[16 + e]; }
    return f.v;
}

// B fragment from pre-shuffled weights: [(kk*numNT + nt)*32 + lane] * 16 contiguous bf16
__device__ __forceinline__ v16bf load_b_frag(const unsigned short* wf, int kk, int nt,
                                             int numNT, int lane) {
    const unsigned short* p = wf + ((((kk * numNT) + nt) * 32 + lane) << 4);
    return *(const v16bf*)p;   // 32B aligned -> 2x b128 loads
}

// Pre-shuffle a row-major (K x N) f32 weight matrix into B-fragment order, bf16.
__global__ void pack_w(const float* __restrict__ W, unsigned short* __restrict__ out,
                       int K, int N) {
    int idx = blockIdx.x * blockDim.x + threadIdx.x;
    if (idx >= K * N) return;
    int k = idx / N, n = idx - k * N;
    int kk = k >> 5, kr = k & 31, nt = n >> 4, nr = n & 15;
    int lane = (((kr >> 3) & 1) << 4) | nr;
    int elem = (kr & 7) | ((kr >> 4) << 3);
    out[((((kk * (N >> 4)) + nt) * 32 + lane) << 4) + elem] = f2bf(W[idx]);
}

// ---------------- Edge MLP + scatter-add ----------------
// One wave per 32-edge tile (two 16-row WMMA tiles sharing every B fragment):
//   h = [x[row] | edge_attr]          (32 x 256)
//   t = ELU(h @ W1a + b1a)            (32 x 256)
//   o = t @ W1b + b1b  -> atomicAdd into agg[col], atomicAdd 1 into cnt[col]
__global__ void __launch_bounds__(32)
edge_mlp_scatter(const float* __restrict__ x, const int* __restrict__ eidx,
                 const float* __restrict__ eattr,
                 const float* __restrict__ b1a, const float* __restrict__ b1b,
                 const unsigned short* __restrict__ W1af,
                 const unsigned short* __restrict__ W1bf,
                 float* __restrict__ agg, float* __restrict__ cnt, int E) {
    __shared__ unsigned short h[32 * 256];   // 16 KB
    __shared__ unsigned short t[32 * 256];   // 16 KB
    __shared__ int srow[32];
    __shared__ int scol[32];

    const int lane = threadIdx.x & 31;
    const int e0   = blockIdx.x * 32;
    const int* rowp = eidx;
    const int* colp = eidx + E;

    {   // one edge per lane
        int e = e0 + lane;
        if (e < E) {
            srow[lane] = rowp[e];
            int c = colp[e];
            scol[lane] = c;
            atomicAdd(&cnt[c], 1.0f);
        } else {
            srow[lane] = 0;
            scol[lane] = -1;
        }
    }
    __syncthreads();

    // Build h: 32 rows x 256 cols, float4 loads -> packed bf16x4 LDS stores.
    // 4 consecutive k stay within one row and one side of the 128-col concat split.
    for (int i = lane * 4; i < 32 * 256; i += 32 * 4) {
        int m = i >> 8;
        int k = i & 255;
        int e = e0 + m;
        float4 v4;
        if (e < E) {
            if (k < 128) v4 = *(const float4*)&x[srow[m] * 128 + k];
            else         v4 = *(const float4*)&eattr[e * 128 + (k - 128)];
        } else {
            v4 = make_float4(0.f, 0.f, 0.f, 0.f);
        }
        uint2 p;
        p.x = pack2bf(v4.x, v4.y);
        p.y = pack2bf(v4.z, v4.w);
        *(uint2*)&h[i] = p;
    }
    __syncthreads();

    const int n  = lane & 15;
    const int mb = (lane >> 4) * 8;

    // Layer 1: 16 N-tiles x 8 K-steps, two M-tiles share each B fragment
    for (int nt = 0; nt < 16; ++nt) {
        float bias = b1a[nt * 16 + n];
        v8f c0, c1;
#pragma unroll
        for (int v = 0; v < 8; ++v) { c0[v] = bias; c1[v] = bias; }
#pragma unroll
        for (int kk = 0; kk < 8; ++kk) {
            v16bf b  = load_b_frag(W1af, kk, nt, 16, lane);
            v16bf a0 = load_a_lds(h,            256, kk * 32, lane);
            v16bf a1 = load_a_lds(h + 16 * 256, 256, kk * 32, lane);
            c0 = __builtin_amdgcn_wmma_f32_16x16x32_bf16(false, a0, false, b,
                                                         (short)0, c0, false, false);
            c1 = __builtin_amdgcn_wmma_f32_16x16x32_bf16(false, a1, false, b,
                                                         (short)0, c1, false, false);
        }
#pragma unroll
        for (int v = 0; v < 8; ++v) {
            t[(mb + v) * 256      + nt * 16 + n] = f2bf(elu1(c0[v]));
            t[(mb + v + 16) * 256 + nt * 16 + n] = f2bf(elu1(c1[v]));
        }
    }
    __syncthreads();

    // Layer 2 + scatter
    for (int nt = 0; nt < 16; ++nt) {
        float bias = b1b[nt * 16 + n];
        v8f c0, c1;
#pragma unroll
        for (int v = 0; v < 8; ++v) { c0[v] = bias; c1[v] = bias; }
#pragma unroll
        for (int kk = 0; kk < 8; ++kk) {
            v16bf b  = load_b_frag(W1bf, kk, nt, 16, lane);
            v16bf a0 = load_a_lds(t,            256, kk * 32, lane);
            v16bf a1 = load_a_lds(t + 16 * 256, 256, kk * 32, lane);
            c0 = __builtin_amdgcn_wmma_f32_16x16x32_bf16(false, a0, false, b,
                                                         (short)0, c0, false, false);
            c1 = __builtin_amdgcn_wmma_f32_16x16x32_bf16(false, a1, false, b,
                                                         (short)0, c1, false, false);
        }
#pragma unroll
        for (int v = 0; v < 8; ++v) {
            int col0 = scol[mb + v];
            int col1 = scol[mb + v + 16];
            if (col0 >= 0) atomicAdd(&agg[col0 * 256 + nt * 16 + n], c0[v]);
            if (col1 >= 0) atomicAdd(&agg[col1 * 256 + nt * 16 + n], c1[v]);
        }
    }
}

// ---------------- Node MLP ----------------
// z = [x | agg/max(cnt,1)] (16x384) ; t = ELU(z@W2a+b2a) (16x256) ; out = t@W2b+b2b (16x128)
__global__ void __launch_bounds__(64)
node_mlp(const float* __restrict__ x, const float* __restrict__ agg,
         const float* __restrict__ cnt,
         const float* __restrict__ b2a, const float* __restrict__ b2b,
         const unsigned short* __restrict__ W2af,
         const unsigned short* __restrict__ W2bf,
         float* __restrict__ out, int N) {
    __shared__ unsigned short sz[2][16 * 384];
    __shared__ unsigned short st[2][16 * 256];
    __shared__ float sinv[2][16];

    const int wave = threadIdx.x >> 5;
    const int lane = threadIdx.x & 31;
    const int tile = blockIdx.x * 2 + wave;
    const int n0   = tile * 16;
    const bool active = (n0 < N);
    unsigned short* z = sz[wave];
    unsigned short* t = st[wave];

    if (active && lane < 16) {
        int node = n0 + lane; if (node >= N) node = 0;
        sinv[wave][lane] = 1.0f / fmaxf(cnt[node], 1.0f);
    }
    __syncthreads();

    if (active) {
        // 16 rows x 384 cols; 4 consecutive k stay in one row and one side of the split
        for (int i = lane * 4; i < 16 * 384; i += 32 * 4) {
            int m = i / 384;
            int k = i - m * 384;
            int node = n0 + m; if (node >= N) node = 0;
            float4 v4;
            if (k < 128) {
                v4 = *(const float4*)&x[node * 128 + k];
            } else {
                v4 = *(const float4*)&agg[node * 256 + (k - 128)];
                float s = sinv[wave][m];
                v4.x *= s; v4.y *= s; v4.z *= s; v4.w *= s;
            }
            uint2 p;
            p.x = pack2bf(v4.x, v4.y);
            p.y = pack2bf(v4.z, v4.w);
            *(uint2*)&z[i] = p;
        }
    }
    __syncthreads();

    const int n  = lane & 15;
    const int mb = (lane >> 4) * 8;

    if (active) {
        for (int nt = 0; nt < 16; ++nt) {
            float bias = b2a[nt * 16 + n];
            v8f c;
#pragma unroll
            for (int v = 0; v < 8; ++v) c[v] = bias;
#pragma unroll
            for (int kk = 0; kk < 12; ++kk) {
                v16bf a = load_a_lds(z, 384, kk * 32, lane);
                v16bf b = load_b_frag(W2af, kk, nt, 16, lane);
                c = __builtin_amdgcn_wmma_f32_16x16x32_bf16(false, a, false, b,
                                                            (short)0, c, false, false);
            }
#pragma unroll
            for (int v = 0; v < 8; ++v)
                t[(mb + v) * 256 + nt * 16 + n] = f2bf(elu1(c[v]));
        }
    }
    __syncthreads();

    if (active) {
        for (int nt = 0; nt < 8; ++nt) {
            float bias = b2b[nt * 16 + n];
            v8f c;
#pragma unroll
            for (int v = 0; v < 8; ++v) c[v] = bias;
#pragma unroll
            for (int kk = 0; kk < 8; ++kk) {
                v16bf a = load_a_lds(t, 256, kk * 32, lane);
                v16bf b = load_b_frag(W2bf, kk, nt, 8, lane);
                c = __builtin_amdgcn_wmma_f32_16x16x32_bf16(false, a, false, b,
                                                            (short)0, c, false, false);
            }
#pragma unroll
            for (int v = 0; v < 8; ++v) {
                int node = n0 + mb + v;
                if (node < N) out[node * 128 + nt * 16 + n] = c[v];
            }
        }
    }
}

extern "C" void kernel_launch(void* const* d_in, const int* in_sizes, int n_in,
                              void* d_out, int out_size, void* d_ws, size_t ws_size,
                              hipStream_t stream) {
    (void)n_in; (void)out_size; (void)ws_size;
    const float* x     = (const float*)d_in[0];
    const int*   eidx  = (const int*)d_in[1];
    const float* eattr = (const float*)d_in[2];
    // d_in[3]=u, d_in[4]=batch : unused (single graph)
    const float* W1a = (const float*)d_in[5];
    const float* b1a = (const float*)d_in[6];
    const float* W1b = (const float*)d_in[7];
    const float* b1b = (const float*)d_in[8];
    const float* W2a = (const float*)d_in[9];
    const float* b2a = (const float*)d_in[10];
    const float* W2b = (const float*)d_in[11];
    const float* b2b = (const float*)d_in[12];
    float* out = (float*)d_out;

    const int E = in_sizes[1] / 2;
    const int N = in_sizes[0] / 128;

    // workspace layout
    char* ws = (char*)d_ws;
    unsigned short* W1af = (unsigned short*)(ws + 0);        // 256x256 bf16 = 131072 B
    unsigned short* W1bf = (unsigned short*)(ws + 131072);   // 131072 B
    unsigned short* W2af = (unsigned short*)(ws + 262144);   // 384x256 bf16 = 196608 B
    unsigned short* W2bf = (unsigned short*)(ws + 458752);   // 256x128 bf16 = 65536 B
    size_t cnt_off = 524288;
    float* cnt = (float*)(ws + cnt_off);                     // N floats
    size_t agg_off = cnt_off + (((size_t)N * 4 + 255) / 256) * 256;
    float* agg = (float*)(ws + agg_off);                     // N x 256 floats

    hipMemsetAsync(cnt, 0, (size_t)N * sizeof(float), stream);
    hipMemsetAsync(agg, 0, (size_t)N * 256 * sizeof(float), stream);

    pack_w<<<(256 * 256 + 255) / 256, 256, 0, stream>>>(W1a, W1af, 256, 256);
    pack_w<<<(256 * 256 + 255) / 256, 256, 0, stream>>>(W1b, W1bf, 256, 256);
    pack_w<<<(384 * 256 + 255) / 256, 256, 0, stream>>>(W2a, W2af, 384, 256);
    pack_w<<<(256 * 128 + 255) / 256, 256, 0, stream>>>(W2b, W2bf, 256, 128);

    int etiles = (E + 31) / 32;
    edge_mlp_scatter<<<etiles, 32, 0, stream>>>(x, eidx, eattr, b1a, b1b,
                                                W1af, W1bf, agg, cnt, E);

    int ntiles = (N + 15) / 16;
    node_mlp<<<(ntiles + 1) / 2, 64, 0, stream>>>(x, agg, cnt, b2a, b2b,
                                                  W2af, W2bf, out, N);
}
